// QEffPlamoAttention_83854941487156
// MI455X (gfx1250) — matrix operand order repaired and checked by reading, not compiled, not run on hardware
//
#include <hip/hip_runtime.h>

// ---------------- types / helpers ----------------
typedef __attribute__((ext_vector_type(16))) __bf16 v16bf;
typedef __attribute__((ext_vector_type(8)))  float  v8f;
typedef __attribute__((ext_vector_type(4)))  unsigned int v4u;
typedef __attribute__((ext_vector_type(8)))  int v8i_t;
typedef __attribute__((ext_vector_type(4)))  int v4i_t;

union Frag {
    uint4 q[2];   // 32 bytes = 16 bf16
    v16bf v;
};

__device__ __forceinline__ unsigned short f2bf(float f) {
    unsigned int u = __float_as_uint(f);
    unsigned int r = u + 0x7fffu + ((u >> 16) & 1u);   // round-to-nearest-even
    return (unsigned short)(r >> 16);
}
__device__ __forceinline__ float bf2f(unsigned short h) {
    return __uint_as_float(((unsigned int)h) << 16);
}

// ---------------- Tensor Data Mover: 2-D tile load into LDS ----------------
// Builds D# group0/group1 per CDNA5 ISA (CDNA5 ISA):
//  g0: count=1 | lds_addr | global_addr[56:0] | type=2
//  g1: data_size=2B, pad_enable/interval/amount, tensor_dim0/1 = tile dims
//      (tile fully in-bounds), tile_dim0/1, tensor_dim0_stride = row stride.
// Remaining groups zero (2-D tensor). cpol = 0.
// This toolchain exposes the 6-arg builtin: (v4u, v8i, v4i, v4i, v8i, i32).
__device__ __forceinline__ void tdm_load_2d(unsigned lds_off, const void* gptr,
                                            unsigned tile_d0, unsigned tile_d1,
                                            unsigned long long row_stride_elems,
                                            int pad_interval, int pad_amount) {
    unsigned long long ga = (unsigned long long)gptr;
    v4u g0;
    g0.x = 1u;                                            // count = 1 (valid)
    g0.y = lds_off;                                       // LDS byte address
    g0.z = (unsigned)ga;                                  // global addr lo
    g0.w = (unsigned)((ga >> 32) & 0x1FFFFFFu) | (2u << 30);  // addr[56:32] | type=2
    v8i_t g1;
    g1[0] = (1 << 16)                    // data_size = 2 bytes (bf16)
          | (1 << 20)                    // pad_enable
          | (pad_interval << 22)         // DWORDs between pads (encoded)
          | (pad_amount << 25);          // pad DWORDs (encoded)
    g1[1] = (int)((tile_d0 & 0xFFFFu) << 16);                         // tensor_dim0 lo16
    g1[2] = (int)((tile_d0 >> 16) & 0xFFFFu) | (int)((tile_d1 & 0xFFFFu) << 16); // dim0 hi | dim1 lo
    g1[3] = (int)((tile_d1 >> 16) & 0xFFFFu) | (int)(tile_d0 << 16);  // dim1 hi | tile_dim0
    g1[4] = (int)tile_d1;                                             // tile_dim1 (tile_dim2 = 0)
    g1[5] = (int)(row_stride_elems & 0xFFFFFFFFull);                  // dim0_stride lo32
    g1[6] = (int)((row_stride_elems >> 32) & 0xFFFFull);              // dim0_stride hi16
    g1[7] = 0;
    v4i_t z4 = {0, 0, 0, 0};
    v8i_t z8 = {0, 0, 0, 0, 0, 0, 0, 0};
    __builtin_amdgcn_tensor_load_to_lds(g0, g1, z4, z4, z8, 0);
}

// ---------------- fp32 -> bf16 cast ----------------
__global__ void cast_f32_bf16(const float* __restrict__ src,
                              unsigned short* __restrict__ dst, int n) {
    int i = blockIdx.x * blockDim.x + threadIdx.x;
    if (i < n) dst[i] = f2bf(src[i]);
}

// ---------------- RoPE (in place, bf16 buffer [S, nheads*128]) ----------------
__global__ void rope_kernel(unsigned short* __restrict__ buf, int nheads, int rowstride) {
    int idx = blockIdx.x * blockDim.x + threadIdx.x;
    int total = 2048 * nheads * 64;
    if (idx >= total) return;
    int d = idx & 63;
    int t = idx >> 6;
    int h = t % nheads;
    int s = t / nheads;
    float inv = __expf(-(float)d * 0.14391565f);   // 10000^(-d/64)
    float ang = (float)s * inv;
    float c  = cosf(ang);
    float sn = sinf(ang);
    size_t base = (size_t)s * rowstride + h * 128 + d;
    float x1 = bf2f(buf[base]);
    float x2 = bf2f(buf[base + 64]);
    buf[base]      = f2bf(x1 * c - x2 * sn);
    buf[base + 64] = f2bf(x2 * c + x1 * sn);
}

// ---------------- bf16 WMMA GEMM: C[M,N] = A[M,K] * B[N,K]^T ----------------
#define TILE_M 128
#define TILE_N 128
#define TILE_K 64
#define LDT    72   // padded LDS stride (bf16): 64 data + 8 pad = 144B rows

// LDS arena: As at byte 0, Bs at byte AS_BYTES (single array -> offsets are exact)
#define GEMM_AS_OFF 0u
#define GEMM_BS_OFF (unsigned)(TILE_M * LDT * 2)

template <bool OUT_BF16>
__global__ __launch_bounds__(256) void gemm_bf16(const unsigned short* __restrict__ A,
                                                 const unsigned short* __restrict__ B,
                                                 void* __restrict__ Cout,
                                                 int M, int N, int K) {
    __shared__ __align__(16) unsigned short SMEM[2 * TILE_M * LDT];
    unsigned short* As = SMEM;
    unsigned short* Bs = SMEM + TILE_M * LDT;
    (void)M;

    const int tid  = threadIdx.x;
    const int lane = tid & 31;
    const int w    = tid >> 5;     // wave id 0..7
    const int hi   = lane >> 4;
    const int lr   = lane & 15;
    const int m0   = blockIdx.y * TILE_M;
    const int n0   = blockIdx.x * TILE_N;

    v8f acc[8] = {};

    for (int k0 = 0; k0 < K; k0 += TILE_K) {
        // TDM staging: wave0 -> A tile, wave1 -> B tile.
        // Tile rows: 64 bf16 = 32 DWORDs (pad_interval enc 4), pad 4 DWORDs (enc 3) -> stride 72.
        if (w == 0) {
            tdm_load_2d(GEMM_AS_OFF, A + (size_t)m0 * K + k0,
                        TILE_K, TILE_M, (unsigned long long)K, 4, 3);
            __builtin_amdgcn_s_wait_tensorcnt(0);
        } else if (w == 1) {
            tdm_load_2d(GEMM_BS_OFF, B + (size_t)n0 * K + k0,
                        TILE_K, TILE_N, (unsigned long long)K, 4, 3);
            __builtin_amdgcn_s_wait_tensorcnt(0);
        }
        __syncthreads();

        #pragma unroll
        for (int kk = 0; kk < TILE_K; kk += 32) {
            Frag af;  // A fragment (16x32): row = lane&15, K subset split by lane half
            af.q[0] = *(const uint4*)(&As[(w * 16 + lr) * LDT + kk + hi * 8]);
            af.q[1] = *(const uint4*)(&As[(w * 16 + lr) * LDT + kk + 16 + hi * 8]);
            #pragma unroll
            for (int t = 0; t < 8; ++t) {
                Frag bfr;  // B fragment (32x16): col = lane&15, K subset split by lane half (matches A)
                bfr.q[0] = *(const uint4*)(&Bs[(t * 16 + lr) * LDT + kk + hi * 8]);
                bfr.q[1] = *(const uint4*)(&Bs[(t * 16 + lr) * LDT + kk + 16 + hi * 8]);
                acc[t] = __builtin_amdgcn_wmma_f32_16x16x32_bf16(
                    false, af.v, false, bfr.v, (short)0, acc[t], false, false);
            }
        }
        __syncthreads();
    }

    const int row_base = m0 + w * 16 + hi * 8;
    if constexpr (OUT_BF16) {
        unsigned short* C = (unsigned short*)Cout;
        #pragma unroll
        for (int t = 0; t < 8; ++t)
            #pragma unroll
            for (int r = 0; r < 8; ++r)
                C[(size_t)(row_base + r) * N + n0 + t * 16 + lr] = f2bf(acc[t][r]);
    } else {
        float* C = (float*)Cout;
        #pragma unroll
        for (int t = 0; t < 8; ++t)
            #pragma unroll
            for (int r = 0; r < 8; ++r)
                C[(size_t)(row_base + r) * N + n0 + t * 16 + lr] = acc[t][r];
    }
}

// ---------------- flash attention (causal, GQA kv = h % 4) ----------------
#define BM  128
#define BN  64
#define LKK 136   // K tile stride  (128 + 8)
#define LVN 72    // V^T tile stride (64 + 8)
#define LPN 72    // P tile stride   (64 + 8)

// LDS arena offsets (bytes): Ks @ 0, Vs @ 17408, Ps @ 35840
#define ATT_KS_ELEMS (BN * LKK)
#define ATT_VS_ELEMS (128 * LVN)
#define ATT_PS_ELEMS (BM * LPN)
#define ATT_KS_OFF   0u

__global__ __launch_bounds__(256) void attn_kernel(const unsigned short* __restrict__ Q,
                                                   const unsigned short* __restrict__ Kb,
                                                   const unsigned short* __restrict__ Vb,
                                                   unsigned short* __restrict__ O) {
    __shared__ __align__(16) unsigned short SMEM[ATT_KS_ELEMS + ATT_VS_ELEMS + ATT_PS_ELEMS];
    unsigned short* Ks = SMEM;                               // [n][d], stride LKK
    unsigned short* Vs = SMEM + ATT_KS_ELEMS;                // transposed [d][n], stride LVN
    unsigned short* Ps = SMEM + ATT_KS_ELEMS + ATT_VS_ELEMS; // [m][n], stride LPN

    const int tid  = threadIdx.x;
    const int lane = tid & 31;
    const int w    = tid >> 5;
    const int hi   = lane >> 4;
    const int lr   = lane & 15;
    const int q0   = blockIdx.x * BM;
    const int h    = blockIdx.y;
    const int kvh  = h & 3;   // jnp.tile -> kv head = h % NKV

    // Q fragments for this wave's 16 rows, all of D=128 (4 chunks of K=32)
    Frag qf[4];
    {
        const unsigned short* qp = Q + (size_t)(q0 + w * 16 + lr) * 4096 + h * 128;
        #pragma unroll
        for (int c = 0; c < 4; ++c) {
            qf[c].q[0] = *(const uint4*)(qp + c * 32 + hi * 8);
            qf[c].q[1] = *(const uint4*)(qp + c * 32 + 16 + hi * 8);
        }
    }

    v8f o[8] = {};
    float m_i[8], l_i[8];
    #pragma unroll
    for (int r = 0; r < 8; ++r) { m_i[r] = -1e30f; l_i[r] = 0.f; }

    const float scale = 0.08838834764831845f;  // 1/sqrt(128)
    const int jmax = q0 + BM;

    for (int j0 = 0; j0 < jmax; j0 += BN) {
        // K tile via TDM (wave 0): 64 rows x 128 bf16 = 64 DWORDs (enc 5), pad 4 DWORDs (enc 3).
        if (w == 0) {
            tdm_load_2d(ATT_KS_OFF, Kb + (size_t)j0 * 512 + kvh * 128,
                        128u, (unsigned)BN, 512ull, 5, 3);
        }
        // V tile staged transposed by all waves (TDM cannot transpose), overlapping the TDM.
        #pragma unroll
        for (int it = 0; it < 4; ++it) {
            int idx = tid + it * 256;
            int row = idx >> 4;          // 0..63
            int c8  = (idx & 15) * 8;    // 0..120
            union { uint4 q; unsigned short s[8]; } vq;
            vq.q = *(const uint4*)(Vb + (size_t)(j0 + row) * 512 + kvh * 128 + c8);
            #pragma unroll
            for (int e = 0; e < 8; ++e) Vs[(c8 + e) * LVN + row] = vq.s[e];
        }
        if (w == 0) __builtin_amdgcn_s_wait_tensorcnt(0);
        __syncthreads();

        // S strip (16 x 64) = Q . K^T
        v8f sacc[4] = {};
        #pragma unroll
        for (int c = 0; c < 4; ++c) {
            #pragma unroll
            for (int t = 0; t < 4; ++t) {
                Frag kf;
                kf.q[0] = *(const uint4*)(&Ks[(t * 16 + lr) * LKK + c * 32 + hi * 8]);
                kf.q[1] = *(const uint4*)(&Ks[(t * 16 + lr) * LKK + c * 32 + 16 + hi * 8]);
                sacc[t] = __builtin_amdgcn_wmma_f32_16x16x32_bf16(
                    false, qf[c].v, false, kf.v, (short)0, sacc[t], false, false);
            }
        }

        // scale + causal mask (-10000 like the reference)
        const int rowg0 = q0 + w * 16 + hi * 8;
        #pragma unroll
        for (int t = 0; t < 4; ++t) {
            int colg = j0 + t * 16 + lr;
            #pragma unroll
            for (int r = 0; r < 8; ++r) {
                float sv = sacc[t][r] * scale;
                if (colg > rowg0 + r) sv = -10000.f;
                sacc[t][r] = sv;
            }
        }

        // online softmax: each row lives across 16 lanes of a half-wave
        #pragma unroll
        for (int r = 0; r < 8; ++r) {
            float mx = fmaxf(fmaxf(sacc[0][r], sacc[1][r]), fmaxf(sacc[2][r], sacc[3][r]));
            mx = fmaxf(mx, __shfl_xor(mx, 1, 32));
            mx = fmaxf(mx, __shfl_xor(mx, 2, 32));
            mx = fmaxf(mx, __shfl_xor(mx, 4, 32));
            mx = fmaxf(mx, __shfl_xor(mx, 8, 32));
            float m_new = fmaxf(m_i[r], mx);
            float corr  = __expf(m_i[r] - m_new);
            m_i[r] = m_new;
            float psum = 0.f;
            #pragma unroll
            for (int t = 0; t < 4; ++t) {
                float p = __expf(sacc[t][r] - m_new);
                psum += p;
                sacc[t][r] = p;
            }
            psum += __shfl_xor(psum, 1, 32);
            psum += __shfl_xor(psum, 2, 32);
            psum += __shfl_xor(psum, 4, 32);
            psum += __shfl_xor(psum, 8, 32);
            l_i[r] = l_i[r] * corr + psum;
            #pragma unroll
            for (int t = 0; t < 8; ++t) o[t][r] *= corr;
        }

        // P -> LDS (bf16) so it can be re-read in A-fragment layout
        #pragma unroll
        for (int t = 0; t < 4; ++t)
            #pragma unroll
            for (int r = 0; r < 8; ++r)
                Ps[(w * 16 + hi * 8 + r) * LPN + t * 16 + lr] = f2bf(sacc[t][r]);
        __syncthreads();

        // O += P . V   (K dim = 64, two WMMA chunks)
        #pragma unroll
        for (int kk = 0; kk < BN; kk += 32) {
            Frag pf;
            pf.q[0] = *(const uint4*)(&Ps[(w * 16 + lr) * LPN + kk + hi * 8]);
            pf.q[1] = *(const uint4*)(&Ps[(w * 16 + lr) * LPN + kk + 16 + hi * 8]);
            #pragma unroll
            for (int t = 0; t < 8; ++t) {
                Frag vf;
                vf.q[0] = *(const uint4*)(&Vs[(t * 16 + lr) * LVN + kk + hi * 8]);
                vf.q[1] = *(const uint4*)(&Vs[(t * 16 + lr) * LVN + kk + 16 + hi * 8]);
                o[t] = __builtin_amdgcn_wmma_f32_16x16x32_bf16(
                    false, pf.v, false, vf.v, (short)0, o[t], false, false);
            }
        }
        __syncthreads();
    }

    // normalize and store bf16 attention output [S, 32*128]
    const int orow0 = q0 + w * 16 + hi * 8;
    #pragma unroll
    for (int t = 0; t < 8; ++t)
        #pragma unroll
        for (int r = 0; r < 8; ++r)
            O[(size_t)(orow0 + r) * 4096 + h * 128 + t * 16 + lr] = f2bf(o[t][r] / l_i[r]);
}

// ---------------- host launcher ----------------
extern "C" void kernel_launch(void* const* d_in, const int* in_sizes, int n_in,
                              void* d_out, int out_size, void* d_ws, size_t ws_size,
                              hipStream_t stream) {
    (void)in_sizes; (void)n_in; (void)out_size; (void)ws_size;
    const float* X  = (const float*)d_in[0];   // hidden_states [2048,4096]
    const float* Wq = (const float*)d_in[3];   // [4096,4096]
    const float* Wk = (const float*)d_in[4];   // [512,4096]
    const float* Wv = (const float*)d_in[5];   // [512,4096]
    const float* Wo = (const float*)d_in[6];   // [4096,4096] (H, NQ*D)
    float* out = (float*)d_out;

    char* ws = (char*)d_ws;
    size_t off = 0;
    auto take = [&](size_t elems) {
        unsigned short* p = (unsigned short*)(ws + off);
        off = (off + elems * 2 + 255) & ~(size_t)255;
        return p;
    };
    unsigned short* Xb  = take((size_t)2048 * 4096);  // reused as attention output later
    unsigned short* Wb  = take((size_t)4096 * 4096);  // holds Wq, then reused for Wo
    unsigned short* Wkb = take((size_t)512 * 4096);
    unsigned short* Wvb = take((size_t)512 * 4096);
    unsigned short* Qb  = take((size_t)2048 * 4096);
    unsigned short* Kb  = take((size_t)2048 * 512);
    unsigned short* Vb  = take((size_t)2048 * 512);

    const int TPB = 256;
    cast_f32_bf16<<<(2048 * 4096 + TPB - 1) / TPB, TPB, 0, stream>>>(X, Xb, 2048 * 4096);
    cast_f32_bf16<<<(4096 * 4096 + TPB - 1) / TPB, TPB, 0, stream>>>(Wq, Wb, 4096 * 4096);
    cast_f32_bf16<<<(512 * 4096 + TPB - 1) / TPB, TPB, 0, stream>>>(Wk, Wkb, 512 * 4096);
    cast_f32_bf16<<<(512 * 4096 + TPB - 1) / TPB, TPB, 0, stream>>>(Wv, Wvb, 512 * 4096);

    gemm_bf16<true><<<dim3(4096 / 128, 2048 / 128), 256, 0, stream>>>(Xb, Wb,  Qb, 2048, 4096, 4096);
    gemm_bf16<true><<<dim3(512 / 128,  2048 / 128), 256, 0, stream>>>(Xb, Wkb, Kb, 2048, 512,  4096);
    gemm_bf16<true><<<dim3(512 / 128,  2048 / 128), 256, 0, stream>>>(Xb, Wvb, Vb, 2048, 512,  4096);

    rope_kernel<<<(2048 * 32 * 64 + TPB - 1) / TPB, TPB, 0, stream>>>(Qb, 32, 4096);
    rope_kernel<<<(2048 * 4 * 64 + TPB - 1) / TPB, TPB, 0, stream>>>(Kb, 4, 512);

    unsigned short* Ob = Xb;  // X no longer needed
    attn_kernel<<<dim3(2048 / 128, 32), 256, 0, stream>>>(Qb, Kb, Vb, Ob);

    cast_f32_bf16<<<(4096 * 4096 + TPB - 1) / TPB, TPB, 0, stream>>>(Wo, Wb, 4096 * 4096);
    gemm_bf16<false><<<dim3(4096 / 128, 2048 / 128), 256, 0, stream>>>(Ob, Wb, out, 2048, 4096, 4096);
}
